// BiLSTM_CRF_84207128805837
// MI455X (gfx1250) — matrix operand order repaired and checked by reading, not compiled, not run on hardware
//
#include <hip/hip_runtime.h>
#include <hip/hip_bf16.h>

// ---------------- problem constants ----------------
#define BATCH 32
#define TLEN  256
#define BT    (BATCH * TLEN)      // 8192
#define EMB   300
#define KPAD  320                 // EMB padded to multiple of 32
#define HID   256
#define G4    1024                // 4*HID
#define TAGS  9
#define NPADC 16                  // TAGS padded to 16

typedef _Float16 v16h_t __attribute__((ext_vector_type(16)));
typedef _Float16 v8h_t  __attribute__((ext_vector_type(8)));
typedef float    v8f_t  __attribute__((ext_vector_type(8)));
typedef int      v4i_t  __attribute__((ext_vector_type(4)));

// ---------------- CDNA5 async global->LDS support (guarded) ----------------
#if defined(__has_builtin)
#  if __has_builtin(__builtin_amdgcn_global_load_async_to_lds_b128)
#    define HAVE_ASYNC_LDS 1
#  endif
#  if __has_builtin(__builtin_amdgcn_s_wait_asynccnt)
#    define HAVE_ASYNC_WAIT 1
#  endif
#endif

typedef __attribute__((address_space(1))) v4i_t* as1_v4i_p;
typedef __attribute__((address_space(3))) v4i_t* as3_v4i_p;

// Copy 16 bytes global -> LDS. Async (ASYNCcnt-tracked) when available.
__device__ __forceinline__ void async_copy16(const _Float16* g, _Float16* l) {
#if defined(HAVE_ASYNC_LDS)
    __builtin_amdgcn_global_load_async_to_lds_b128(
        (as1_v4i_p)(uintptr_t)g,   // global AS(1) pointer to int4 (drops const)
        (as3_v4i_p)l,              // addrspacecast generic -> LDS AS(3)
        0, 0);
#else
    *(v8h_t*)l = *(const v8h_t*)g;
#endif
}

__device__ __forceinline__ void async_wait0() {
#if defined(HAVE_ASYNC_LDS) && defined(HAVE_ASYNC_WAIT)
    __builtin_amdgcn_s_wait_asynccnt(0);
#endif
}

// ---------------- workspace layout (bytes) ----------------
static constexpr size_t SZ_X      = (size_t)BT * KPAD * 2;     // f16 embeddings, padded
static constexpr size_t SZ_WIHT   = (size_t)KPAD * G4 * 2;     // f16 Wih.T (320 x 1024)
static constexpr size_t SZ_WHHT   = (size_t)HID * G4 * 2;      // f16 Whh.T (256 x 1024)
static constexpr size_t SZ_WCT    = (size_t)(2 * HID) * NPADC * 2; // f16 Wc.T (512 x 16)
static constexpr size_t SZ_GPRE   = (size_t)BT * G4 * 2;       // f16 input-proj gates
static constexpr size_t SZ_SEQ    = (size_t)BT * (2 * HID) * 2;// f16 [h_f | h_b]

static constexpr size_t OFF_X      = 0;
static constexpr size_t OFF_WIHT_F = OFF_X + SZ_X;
static constexpr size_t OFF_WIHT_B = OFF_WIHT_F + SZ_WIHT;
static constexpr size_t OFF_WHHT_F = OFF_WIHT_B + SZ_WIHT;
static constexpr size_t OFF_WHHT_B = OFF_WHHT_F + SZ_WHHT;
static constexpr size_t OFF_WCT    = OFF_WHHT_B + SZ_WHHT;
static constexpr size_t OFF_GPRE_F = (OFF_WCT + SZ_WCT + 255) & ~(size_t)255;
static constexpr size_t OFF_GPRE_B = OFF_GPRE_F + SZ_GPRE;
static constexpr size_t OFF_SEQ    = OFF_GPRE_B + SZ_GPRE;

// ---------------- WMMA helpers ----------------
__device__ __forceinline__ v8f_t wmma_f16(v16h_t a, v16h_t b, v8f_t c) {
    return __builtin_amdgcn_wmma_f32_16x16x32_f16(
        /*neg_a=*/false, a, /*neg_b=*/false, b,
        /*c_mod=*/(short)0, c, /*reuse_a=*/false, /*reuse_b=*/false);
}

// A-matrix fragment (16x32 f16). Per ISA: lanes 0-15 -> M=lane, k in {0..7}u{16..23};
// lanes 16-31 -> M=lane-16, k in {8..15}u{24..31}.
__device__ __forceinline__ v16h_t load_a_frag(const _Float16* base, int ldk,
                                              int m0, int k0, int lane) {
    int row = m0 + (lane & 15);
    int kb  = k0 + ((lane >> 4) << 3);
    const _Float16* p = base + (size_t)row * ldk + kb;
    v8h_t lo = *(const v8h_t*)(p);
    v8h_t hi = *(const v8h_t*)(p + 16);
    v16h_t r;
#pragma unroll
    for (int i = 0; i < 8; ++i) { r[i] = lo[i]; r[i + 8] = hi[i]; }
    return r;
}

// B-matrix fragment (32x16 f16). lane L holds column N = n0+L%16, k in {8h..8h+7}u{16+8h..16+8h+7}, h=L/16.
__device__ __forceinline__ v16h_t load_b_frag(const _Float16* base, int ldn,
                                              int k0, int n0, int lane) {
    int col = n0 + (lane & 15);
    int kb  = k0 + ((lane >> 4) << 3);
    v16h_t r;
#pragma unroll
    for (int i = 0; i < 8; ++i) {
        r[i]     = base[(size_t)(kb + i) * ldn + col];
        r[i + 8] = base[(size_t)(kb + 16 + i) * ldn + col];
    }
    return r;
}

__device__ __forceinline__ float sigf(float x) { return 1.0f / (1.0f + __expf(-x)); }

// ---------------- prep kernels ----------------
// Generic transpose+pad+convert: W (N x K) f32 row-major -> WT (KPAD x NPAD) f16
__global__ void k_transpose_w(const float* __restrict__ W, _Float16* __restrict__ WT,
                              int N, int K, int Kpad, int Npad) {
    long i = (long)blockIdx.x * blockDim.x + threadIdx.x;
    long total = (long)Kpad * Npad;
    if (i >= total) return;
    int k = (int)(i / Npad);
    int n = (int)(i % Npad);
    float v = (k < K && n < N) ? W[(size_t)n * K + k] : 0.0f;
    WT[i] = (_Float16)v;
}

// Embedding gather + f16 convert + K padding: X (BT x KPAD)
__global__ void k_embed(const int* __restrict__ unigrams, const float* __restrict__ emb,
                        _Float16* __restrict__ X) {
    long i = (long)blockIdx.x * blockDim.x + threadIdx.x;
    if (i >= (long)BT * KPAD) return;
    int row = (int)(i / KPAD);
    int k   = (int)(i % KPAD);
    float v = 0.0f;
    if (k < EMB) {
        int tok = unigrams[row];
        v = emb[(size_t)tok * EMB + k];
    }
    X[i] = (_Float16)v;
}

// ---------------- input projection GEMM (both directions) ----------------
// Gpre_d[BT x 1024] = X[BT x 320] @ WihT_d[320 x 1024] + (bih_d + bhh_d)
// 65536 tiles total (2 dirs * 512 mTiles * 64 nTiles); 8 waves/block, exact grid.
__global__ void __launch_bounds__(256) k_in_gemm(
        const _Float16* __restrict__ X,
        const _Float16* __restrict__ WihT_f, const _Float16* __restrict__ WihT_b,
        const float* __restrict__ bih_f, const float* __restrict__ bhh_f,
        const float* __restrict__ bih_b, const float* __restrict__ bhh_b,
        _Float16* __restrict__ Gf, _Float16* __restrict__ Gb) {
    int lane = threadIdx.x & 31;
    int wid  = threadIdx.x >> 5;
    int tile = blockIdx.x * 8 + wid;          // 0 .. 65535
    int dir   = tile >> 15;
    int t2    = tile & 32767;
    int mTile = t2 >> 6;                      // 0 .. 511
    int nTile = t2 & 63;                      // 0 .. 63

    const _Float16* Wt = dir ? WihT_b : WihT_f;
    const float* b1 = dir ? bih_b : bih_f;
    const float* b2 = dir ? bhh_b : bhh_f;
    _Float16* Gp = dir ? Gb : Gf;

    v8f_t acc = {};
#pragma unroll
    for (int kt = 0; kt < KPAD / 32; ++kt) {
        v16h_t a = load_a_frag(X, KPAD, mTile * 16, kt * 32, lane);
        v16h_t b = load_b_frag(Wt, G4, kt * 32, nTile * 16, lane);
        acc = wmma_f16(a, b, acc);
    }
    int n = nTile * 16 + (lane & 15);
    float bias = b1[n] + b2[n];
    int moff = (lane >> 4) << 3;
#pragma unroll
    for (int r = 0; r < 8; ++r) {
        int m = mTile * 16 + r + moff;
        Gp[(size_t)m * G4 + n] = (_Float16)(acc[r] + bias);
    }
}

// ---------------- recurrent LSTM (persistent, 1 block per direction) ----------------
// Double-buffered LDS staging of Gpre[t] via async global->LDS copies: the next
// step's 64KB gate-preactivation slice streams into LDS (ASYNCcnt) while the
// current step's WMMA GEMM + pointwise run. The staging buffer doubles as the
// gate accumulator: each (m,n) is read (pre) then overwritten (acc+pre) by the
// same lane, so no extra LDS is needed. LDS total: 16 + 32 + 128 = 176 KB.
__global__ void __launch_bounds__(1024) k_lstm(
        const _Float16* __restrict__ Gpre_f, const _Float16* __restrict__ Gpre_b,
        const _Float16* __restrict__ WhhT_f, const _Float16* __restrict__ WhhT_b,
        _Float16* __restrict__ seqout) {
    __shared__ __align__(32) _Float16 sh_h[BATCH * HID];      // 16 KB
    __shared__ __align__(32) float    sh_c[BATCH * HID];      // 32 KB
    __shared__ __align__(32) _Float16 sh_g[2][BATCH * G4];    // 2 x 64 KB

    int dir = blockIdx.x;
    const _Float16* Gpre = dir ? Gpre_b : Gpre_f;
    const _Float16* Wt   = dir ? WhhT_b : WhhT_f;

    int tid  = threadIdx.x;
    int lane = tid & 31;
    int wid  = tid >> 5;                       // 0 .. 31

    // Staging assignment: row m = tid/32, 64-byte chunk (tid%32) of that row,
    // 4 x b128 per thread covers the 64 KB slice with 1024 threads.
    int srow = tid >> 5;                       // 0 .. 31
    int scol = (tid & 31) * 32;                // halfs (=64B) within the 1024-wide row

    for (int i = tid; i < BATCH * HID; i += 1024) {
        sh_h[i] = (_Float16)0.0f;
        sh_c[i] = 0.0f;
    }
    {   // prologue: stage Gpre[t0] into buffer 0
        int t0 = dir ? (TLEN - 1) : 0;
        const _Float16* gsrc = Gpre + ((size_t)srow * TLEN + t0) * G4 + scol;
        _Float16* ldst = &sh_g[0][srow * G4 + scol];
#pragma unroll
        for (int k2 = 0; k2 < 4; ++k2) async_copy16(gsrc + k2 * 8, ldst + k2 * 8);
    }
    async_wait0();
    __syncthreads();

    for (int step = 0; step < TLEN; ++step) {
        int t   = dir ? (TLEN - 1 - step) : step;
        int cur = step & 1;
        _Float16* gcur = &sh_g[cur][0];

        // kick off async staging of the NEXT step's Gpre slice (uniform branch)
        if (step + 1 < TLEN) {
            int tn = dir ? (TLEN - 2 - step) : (step + 1);
            const _Float16* gsrc = Gpre + ((size_t)srow * TLEN + tn) * G4 + scol;
            _Float16* ldst = &sh_g[cur ^ 1][srow * G4 + scol];
#pragma unroll
            for (int k2 = 0; k2 < 4; ++k2) async_copy16(gsrc + k2 * 8, ldst + k2 * 8);
        }

        // GEMM: gates(32x1024) = h(32x256) @ WhhT(256x1024) + staged Gpre[t]
        // 128 tiles (2 mTiles x 64 nTiles), 4 per wave.
#pragma unroll
        for (int q = 0; q < 4; ++q) {
            int tile  = wid * 4 + q;
            int mTile = tile >> 6;
            int nTile = tile & 63;
            v8f_t acc = {};
#pragma unroll
            for (int kt = 0; kt < HID / 32; ++kt) {
                v16h_t a = load_a_frag(sh_h, HID, mTile * 16, kt * 32, lane);
                v16h_t b = load_b_frag(Wt, G4, kt * 32, nTile * 16, lane);
                acc = wmma_f16(a, b, acc);
            }
            int n = nTile * 16 + (lane & 15);
            int moff = (lane >> 4) << 3;
#pragma unroll
            for (int r = 0; r < 8; ++r) {
                int m = mTile * 16 + r + moff;                 // batch index
                float pre = (float)gcur[m * G4 + n];           // staged in LDS
                gcur[m * G4 + n] = (_Float16)(acc[r] + pre);
            }
        }
        __syncthreads();

        // pointwise LSTM cell: 8192 (m,j) pairs over 1024 threads
#pragma unroll
        for (int p = 0; p < 8; ++p) {
            int idx = tid + p * 1024;
            int m = idx >> 8;          // / HID
            int j = idx & 255;         // % HID
            float gi = (float)gcur[m * G4 + j];
            float gf = (float)gcur[m * G4 + HID + j];
            float gg = (float)gcur[m * G4 + 2 * HID + j];
            float go = (float)gcur[m * G4 + 3 * HID + j];
            float c  = sigf(gf) * sh_c[m * HID + j] + sigf(gi) * tanhf(gg);
            float h  = sigf(go) * tanhf(c);
            sh_c[m * HID + j] = c;
            _Float16 hh = (_Float16)h;
            sh_h[m * HID + j] = hh;
            seqout[((size_t)m * TLEN + t) * (2 * HID) + dir * HID + j] = hh;
        }
        async_wait0();     // next step's Gpre slice fully landed in LDS
        __syncthreads();
    }
}

// ---------------- output projection ----------------
// tag_scores(8192 x 9) = seqout(8192 x 512) @ WcT(512 x 16) + bc  (cols >= 9 dropped)
// 512 mTiles, 8 waves/block, 64 blocks exact.
__global__ void __launch_bounds__(256) k_out_gemm(
        const _Float16* __restrict__ seqout, const _Float16* __restrict__ WcT,
        const float* __restrict__ bc, float* __restrict__ out) {
    int lane  = threadIdx.x & 31;
    int mTile = blockIdx.x * 8 + (threadIdx.x >> 5);
    v8f_t acc = {};
#pragma unroll
    for (int kt = 0; kt < (2 * HID) / 32; ++kt) {
        v16h_t a = load_a_frag(seqout, 2 * HID, mTile * 16, kt * 32, lane);
        v16h_t b = load_b_frag(WcT, NPADC, kt * 32, 0, lane);
        acc = wmma_f16(a, b, acc);
    }
    int n = lane & 15;
    int moff = (lane >> 4) << 3;
    if (n < TAGS) {
        float bias = bc[n];
#pragma unroll
        for (int r = 0; r < 8; ++r) {
            int m = mTile * 16 + r + moff;
            out[(size_t)m * TAGS + n] = acc[r] + bias;
        }
    }
}

// ---------------- CRF negative log-likelihood ----------------
// One wave; lane = batch element. emissions are the f32 tag_scores in d_out.
__global__ void __launch_bounds__(32) k_crf(
        const float* __restrict__ em, const int* __restrict__ tags,
        const int* __restrict__ mask, const float* __restrict__ startT,
        const float* __restrict__ endT, const float* __restrict__ trans,
        float* __restrict__ loss_out) {
    __shared__ float s_trans[TAGS * TAGS];
    __shared__ float s_red[32];
    int lane = threadIdx.x;
    for (int i = lane; i < TAGS * TAGS; i += 32) s_trans[i] = trans[i];
    __syncthreads();

    const float* E  = em   + (size_t)lane * TLEN * TAGS;
    const int*   tg = tags + lane * TLEN;
    const int*   mk = mask + lane * TLEN;

    int t0 = tg[0];
    float score = startT[t0] + E[t0];
    int prev = t0;
    int last = t0;

    float alpha[TAGS];
#pragma unroll
    for (int j = 0; j < TAGS; ++j) alpha[j] = startT[j] + E[j];

    for (int t = 1; t < TLEN; ++t) {
        int cur = tg[t];
        int mt  = mk[t];
        if (mt) {
            score += s_trans[prev * TAGS + cur] + E[t * TAGS + cur];
            last = cur;
        }
        prev = cur;
        // forward recursion: next[jn] = logsumexp_ji(alpha[ji] + trans[ji][jn]) + E[t][jn]
        float na[TAGS];
#pragma unroll
        for (int jn = 0; jn < TAGS; ++jn) {
            float mx = -3.0e38f;
#pragma unroll
            for (int ji = 0; ji < TAGS; ++ji)
                mx = fmaxf(mx, alpha[ji] + s_trans[ji * TAGS + jn]);
            float s = 0.0f;
#pragma unroll
            for (int ji = 0; ji < TAGS; ++ji)
                s += __expf(alpha[ji] + s_trans[ji * TAGS + jn] - mx);
            na[jn] = mx + __logf(s) + E[t * TAGS + jn];
        }
        if (mt) {
#pragma unroll
            for (int j = 0; j < TAGS; ++j) alpha[j] = na[j];
        }
    }
    score += endT[last];
    float mx = -3.0e38f;
#pragma unroll
    for (int j = 0; j < TAGS; ++j) mx = fmaxf(mx, alpha[j] + endT[j]);
    float s = 0.0f;
#pragma unroll
    for (int j = 0; j < TAGS; ++j) s += __expf(alpha[j] + endT[j] - mx);
    float denom = mx + __logf(s);

    s_red[lane] = score - denom;
    __syncthreads();
    if (lane == 0) {
        float tot = 0.0f;
        for (int i = 0; i < 32; ++i) tot += s_red[i];
        loss_out[0] = -tot;     // loss = -llh
    }
}

// ---------------- host launcher ----------------
extern "C" void kernel_launch(void* const* d_in, const int* in_sizes, int n_in,
                              void* d_out, int out_size, void* d_ws, size_t ws_size,
                              hipStream_t stream) {
    (void)in_sizes; (void)n_in; (void)out_size; (void)ws_size;
    const int*   unigrams  = (const int*)  d_in[0];
    const int*   input_mask= (const int*)  d_in[1];
    const int*   input_tags= (const int*)  d_in[2];
    const float* emb_table = (const float*)d_in[3];
    const float* Wih_f     = (const float*)d_in[4];
    const float* Whh_f     = (const float*)d_in[5];
    const float* bih_f     = (const float*)d_in[6];
    const float* bhh_f     = (const float*)d_in[7];
    const float* Wih_b     = (const float*)d_in[8];
    const float* Whh_b     = (const float*)d_in[9];
    const float* bih_b     = (const float*)d_in[10];
    const float* bhh_b     = (const float*)d_in[11];
    const float* Wc        = (const float*)d_in[12];
    const float* bc        = (const float*)d_in[13];
    const float* start_tr  = (const float*)d_in[14];
    const float* end_tr    = (const float*)d_in[15];
    const float* trans     = (const float*)d_in[16];

    char* ws = (char*)d_ws;
    _Float16* X       = (_Float16*)(ws + OFF_X);
    _Float16* WihT_f  = (_Float16*)(ws + OFF_WIHT_F);
    _Float16* WihT_b  = (_Float16*)(ws + OFF_WIHT_B);
    _Float16* WhhT_f  = (_Float16*)(ws + OFF_WHHT_F);
    _Float16* WhhT_b  = (_Float16*)(ws + OFF_WHHT_B);
    _Float16* WcT     = (_Float16*)(ws + OFF_WCT);
    _Float16* Gpre_f  = (_Float16*)(ws + OFF_GPRE_F);
    _Float16* Gpre_b  = (_Float16*)(ws + OFF_GPRE_B);
    _Float16* seqout  = (_Float16*)(ws + OFF_SEQ);

    float* tag_scores = (float*)d_out;             // 8192 * 9
    float* loss       = (float*)d_out + BT * TAGS; // 1

    // weight prep
    {
        long tot = (long)KPAD * G4;
        k_transpose_w<<<(tot + 255) / 256, 256, 0, stream>>>(Wih_f, WihT_f, G4, EMB, KPAD, G4);
        k_transpose_w<<<(tot + 255) / 256, 256, 0, stream>>>(Wih_b, WihT_b, G4, EMB, KPAD, G4);
    }
    {
        long tot = (long)HID * G4;
        k_transpose_w<<<(tot + 255) / 256, 256, 0, stream>>>(Whh_f, WhhT_f, G4, HID, HID, G4);
        k_transpose_w<<<(tot + 255) / 256, 256, 0, stream>>>(Whh_b, WhhT_b, G4, HID, HID, G4);
    }
    {
        long tot = (long)(2 * HID) * NPADC;
        k_transpose_w<<<(tot + 255) / 256, 256, 0, stream>>>(Wc, WcT, TAGS, 2 * HID, 2 * HID, NPADC);
    }
    // embedding gather
    {
        long tot = (long)BT * KPAD;
        k_embed<<<(tot + 255) / 256, 256, 0, stream>>>(unigrams, emb_table, X);
    }
    // input projection, both directions: 65536 tiles / 8 waves = 8192 blocks
    k_in_gemm<<<8192, 256, 0, stream>>>(X, WihT_f, WihT_b,
                                        bih_f, bhh_f, bih_b, bhh_b, Gpre_f, Gpre_b);
    // recurrence: 1 block per direction
    k_lstm<<<2, 1024, 0, stream>>>(Gpre_f, Gpre_b, WhhT_f, WhhT_b, seqout);
    // output projection: 512 mTiles / 8 waves = 64 blocks
    k_out_gemm<<<64, 256, 0, stream>>>(seqout, WcT, bc, tag_scores);
    // CRF loss
    k_crf<<<1, 32, 0, stream>>>(tag_scores, input_tags, input_mask,
                                start_tr, end_tr, trans, loss);
}
